// VocabularyProjectionFFN_45397804318817
// MI455X (gfx1250) — matrix operand order repaired and checked by reading, not compiled, not run on hardware
//
#include <hip/hip_runtime.h>
#include <hip/hip_bf16.h>

// ---------------------------------------------------------------------------
// VocabularyProjectionFFN on gfx1250 (MI455X)
//
// Math:  M[h,v] = sum_d vocab_emb[v, h*32+d]           (32 x 32000)
//        proj(Q) = softmax(Q*M / temp) * M^T           (flash-attention form)
//        out = proj( refine( proj( x@Wffn^T + b ) ) )
//
// Flash kernel: 8 waves/block share one vocab split; M-tiles are staged in
// LDS via double-buffered global_load_async_to_lds_b128 (ASYNCcnt), waves
// read WMMA A-tiles with ds_load_b128. Softmax runs in exp2 domain on raw
// v_exp_f32; cross-lane traffic is a single ds_swizzle SWAPX16.
// ---------------------------------------------------------------------------

typedef _Float16 half8 __attribute__((ext_vector_type(8)));
typedef _Float16 v16h  __attribute__((ext_vector_type(16)));
typedef float    v8f   __attribute__((ext_vector_type(8)));
typedef int      v4i   __attribute__((ext_vector_type(4)));

#define VOCAB     32000
#define NROWS     4096            // (B*T)*N_HEAD = 128*32
#define NSPLIT    8               // flash-decoding vocab splits
#define VSPLIT    (VOCAB / NSPLIT)
#define NBLK      (NROWS / 16)    // 256 row-blocks of 16 rows
#define NTASK     (NBLK * NSPLIT) // 2048 wave-tasks
#define GRID_FLASH (NTASK / 8)    // 8 waves per block
#define PART_F    544             // floats per partial: 512 O + 16 m + 16 l
#define LOG2E     1.44269504088896340736f
#define TEMP_MINV 0.1f

static __device__ __forceinline__ float swap16f(float x) {
  // ds_swizzle SWAPX16: xor=0x10, or=0, and=0x1f  -> offset 0x401f
  return __int_as_float(__builtin_amdgcn_ds_swizzle(__float_as_int(x), 0x401f));
}

static __device__ __forceinline__ float fexp2(float x) {
  // raw v_exp_f32 (exp2); skips OCML's subnormal fixup (fine for softmax)
#if __has_builtin(__builtin_amdgcn_exp2f)
  return __builtin_amdgcn_exp2f(x);
#else
  return exp2f(x);
#endif
}

static __device__ __forceinline__ v16h cat8(half8 a, half8 b) {
  return __builtin_shufflevector(a, b, 0,1,2,3,4,5,6,7,8,9,10,11,12,13,14,15);
}

static __device__ __forceinline__ v8f wmma_f16(v16h a, v16h b, v8f c) {
  // D = A(16x32) x B(32x16) + C, f16 in / f32 accum
  return __builtin_amdgcn_wmma_f32_16x16x32_f16(false, a, false, b,
                                                (short)0, c, false, false);
}

// 16B global -> LDS copy, async when the gfx1250 builtin is available.
// Builtin signature (from hipcc diagnostic): param0 = int4* in global AS(1),
// param1 = int4* in shared AS(3), then imm offset + cpol.
static __device__ __forceinline__ void copy16_async(void* lds, const void* g) {
#if __has_builtin(__builtin_amdgcn_global_load_async_to_lds_b128)
  __builtin_amdgcn_global_load_async_to_lds_b128(
      (__attribute__((address_space(1))) v4i*)(void*)g,
      (__attribute__((address_space(3))) v4i*)lds, 0, 0);
#else
  *(half8*)lds = *(const half8*)g;   // sync fallback (DScnt, covered by barrier)
#endif
}

static __device__ __forceinline__ void wait_async_le1() {
#if __has_builtin(__builtin_amdgcn_global_load_async_to_lds_b128)
#if __has_builtin(__builtin_amdgcn_s_wait_asynccnt)
  __builtin_amdgcn_s_wait_asynccnt(1);
#else
  asm volatile("s_wait_asynccnt 1" ::: "memory");
#endif
#endif
}

static __device__ __forceinline__ void wait_async_le0() {
#if __has_builtin(__builtin_amdgcn_global_load_async_to_lds_b128)
#if __has_builtin(__builtin_amdgcn_s_wait_asynccnt)
  __builtin_amdgcn_s_wait_asynccnt(0);
#else
  asm volatile("s_wait_asynccnt 0" ::: "memory");
#endif
#endif
}

// --------------------------- kernel A: build M -----------------------------
// Mh[v][h] feeds score A-tiles (M^T layout); Mt[h][v] feeds output A-tiles.
__global__ __launch_bounds__(256) void k_reduce_vocab(
    const float* __restrict__ ve, _Float16* __restrict__ Mh,
    _Float16* __restrict__ Mt) {
  int t = blockIdx.x * 256 + threadIdx.x;          // VOCAB*32 threads
  int v = t >> 5, h = t & 31;
  const float4* p = (const float4*)(ve + (size_t)v * 1024 + h * 32);
  float s = 0.f;
  #pragma unroll
  for (int i = 0; i < 8; ++i) { float4 f = p[i]; s += (f.x + f.y) + (f.z + f.w); }
  Mh[(size_t)v * 32 + h]    = (_Float16)s;
  Mt[(size_t)h * VOCAB + v] = (_Float16)s;
}

// --------------------------- kernel B: prep Q ------------------------------
__global__ __launch_bounds__(256) void k_qprep(
    const float* __restrict__ x, const float* __restrict__ W,
    const float* __restrict__ b, const float* __restrict__ temps,
    _Float16* __restrict__ Qf) {
  int t = blockIdx.x * 256 + threadIdx.x;          // NROWS*32 threads
  int row = t >> 5, d = t & 31;
  const float* xr = x + (size_t)row * 32;
  const float* wr = W + d * 32;
  float s = b[d];
  #pragma unroll
  for (int k = 0; k < 32; ++k) s += xr[k] * wr[k];
  float tp = temps[row & 31];
  tp = tp < TEMP_MINV ? TEMP_MINV : tp;
  Qf[(size_t)row * 32 + d] = (_Float16)(s * (LOG2E / tp));
}

// --------------------------- kernel C: flash -------------------------------
__global__ __launch_bounds__(256) void k_flash(
    const _Float16* __restrict__ Mh, const _Float16* __restrict__ Mt,
    const _Float16* __restrict__ Qf, float* __restrict__ part) {
  // double-buffered staged tiles: 32 rows x 32 cols f16 each
  __shared__ __align__(16) _Float16 lmh[2][32 * 32];  // [v_local][h]
  __shared__ __align__(16) _Float16 lmt[2][32 * 32];  // [h][v_local]

  const int tid  = threadIdx.x;
  const int lane = tid & 31;
  const int wave = tid >> 5;
  const int rg   = blockIdx.x / NSPLIT;            // row-group (8 rowblocks)
  const int sp   = blockIdx.x % NSPLIT;            // vocab split
  const int rb   = rg * 8 + wave;                  // this wave's rowblock
  const int task = rb * NSPLIT + sp;
  const int lo16 = lane & 15;
  const int hi   = lane >> 4;

  // B matrix = Q^T (32h x 16r): lanes 0-15 K=0..7,16..23, lanes 16-31 K=8..15,24..31.
  const _Float16* qp = Qf + ((size_t)(rb * 16 + lo16) * 32 + hi * 8);
  v16h qb = cat8(*(const half8*)qp, *(const half8*)(qp + 16));

  // staging: waves 0-3 cover Mh tile, waves 4-7 cover Mt tile; 16B/thread.
  const int half_id = tid & 127;                   // 0..127 within group
  const int srow = half_id >> 2;                   // 0..31
  const int scol = (half_id & 3) * 8;              // 0,8,16,24 halves
  const bool do_mh = tid < 128;

  v8f acc0 = {0,0,0,0,0,0,0,0};
  v8f acc1 = {0,0,0,0,0,0,0,0};
  v8f zero = {0,0,0,0,0,0,0,0};
  float m = -__builtin_inff();
  float l = 0.f;

  const int vbeg = sp * VSPLIT, vend = vbeg + VSPLIT;
  int buf = 0;

  // prologue: stage first tile
  if (do_mh)
    copy16_async(&lmh[0][srow * 32 + scol], Mh + (size_t)(vbeg + srow) * 32 + scol);
  else
    copy16_async(&lmt[0][srow * 32 + scol], Mt + (size_t)srow * VOCAB + vbeg + scol);

  for (int v0 = vbeg; v0 < vend; v0 += 32) {
    const int nxt = v0 + 32;
    if (nxt < vend) {                              // stage next tile
      if (do_mh)
        copy16_async(&lmh[buf ^ 1][srow * 32 + scol],
                     Mh + (size_t)(nxt + srow) * 32 + scol);
      else
        copy16_async(&lmt[buf ^ 1][srow * 32 + scol],
                     Mt + (size_t)srow * VOCAB + nxt + scol);
      wait_async_le1();                            // my current-buffer part done
    } else {
      wait_async_le0();
    }
    __syncthreads();                               // everyone's part done

    // ---- score A-tiles from LDS: A = M^T[16v x 32h]
    const _Float16* s0p = &lmh[buf][(lo16)      * 32 + hi * 8];
    const _Float16* s1p = &lmh[buf][(16 + lo16) * 32 + hi * 8];
    v16h a0 = cat8(*(const half8*)s0p, *(const half8*)(s0p + 16));
    v16h a1 = cat8(*(const half8*)s1p, *(const half8*)(s1p + 16));

    v8f s0 = wmma_f16(a0, qb, zero);               // S^T tile v0..v0+15
    v8f s1 = wmma_f16(a1, qb, zero);               // S^T tile v0+16..v0+31

    // ---- online softmax (exp2 domain); partner lane = lane^16
    float pm = s0[0];
    #pragma unroll
    for (int g = 1; g < 8; ++g) pm = fmaxf(pm, s0[g]);
    #pragma unroll
    for (int g = 0; g < 8; ++g) pm = fmaxf(pm, s1[g]);
    pm = fmaxf(pm, swap16f(pm));
    float mn = fmaxf(m, pm);
    float so = fexp2(m - mn);                      // 0 on first iteration

    float p0[8], p1[8], psum = 0.f;
    #pragma unroll
    for (int g = 0; g < 8; ++g) {
      p0[g] = fexp2(s0[g] - mn);
      p1[g] = fexp2(s1[g] - mn);
      psum += p0[g] + p1[g];
    }
    l = l * so + (psum + swap16f(psum));
    m = mn;
    #pragma unroll
    for (int g = 0; g < 8; ++g) { acc0[g] *= so; acc1[g] *= so; }

    // ---- P^T (32v x 16r) in B layout via one SWAPX16 exchange
    v16h pb;
    #pragma unroll
    for (int g = 0; g < 8; ++g) {
      float pres = hi ? p0[g] : p1[g];
      float rec  = swap16f(pres);
      pb[g]     = (_Float16)(hi ? rec   : p0[g]);
      pb[8 + g] = (_Float16)(hi ? p1[g] : rec);
    }

    // ---- output A-tiles from LDS: A = M[16h x 32v]
    const _Float16* o0p = &lmt[buf][(lo16)      * 32 + hi * 16];
    const _Float16* o1p = &lmt[buf][(16 + lo16) * 32 + hi * 16];
    v16h b0 = cat8(*(const half8*)o0p, *(const half8*)(o0p + 8));
    v16h b1 = cat8(*(const half8*)o1p, *(const half8*)(o1p + 8));

    acc0 = wmma_f16(b0, pb, acc0);                 // O^T h=0..15
    acc1 = wmma_f16(b1, pb, acc1);                 // O^T h=16..31

    __syncthreads();                               // done reading buf
    buf ^= 1;
  }

  // partial: O[16r x 32h] + m[16] + l[16]
  float* base = part + (size_t)task * PART_F;
  #pragma unroll
  for (int g = 0; g < 8; ++g) {
    base[lo16 * 32 + g + 8 * hi]      = acc0[g];
    base[lo16 * 32 + 16 + g + 8 * hi] = acc1[g];
  }
  if (hi == 0) { base[512 + lo16] = m; base[528 + lo16] = l; }
}

// ---------------- merge helper (exp2 domain, NSPLIT partials) --------------
static __device__ __forceinline__ float merge_out(const float* __restrict__ part,
                                                  int rb, int r, int h) {
  float mstar = -__builtin_inff();
  #pragma unroll
  for (int s = 0; s < NSPLIT; ++s)
    mstar = fmaxf(mstar, part[(size_t)(rb * NSPLIT + s) * PART_F + 512 + r]);
  float L = 0.f, O = 0.f;
  #pragma unroll
  for (int s = 0; s < NSPLIT; ++s) {
    const float* b = part + (size_t)(rb * NSPLIT + s) * PART_F;
    float w = fexp2(b[512 + r] - mstar);
    L += b[528 + r] * w;
    O += b[r * 32 + h] * w;
  }
  return O / L;
}

// -------------------- kernel D: merge + refinement + Q2 --------------------
__global__ __launch_bounds__(256) void k_merge_refine(
    const float* __restrict__ part, const float* __restrict__ x,
    const float* __restrict__ Wr, const float* __restrict__ br,
    const float* __restrict__ Wg, const float* __restrict__ bg,
    const float* __restrict__ temps, _Float16* __restrict__ Qf2) {
  int t = blockIdx.x * 256 + threadIdx.x;          // NROWS*32 threads
  int row = t >> 5, h = t & 31;
  float vout = merge_out(part, row >> 4, row & 15, h);

  const float* xr = x + (size_t)row * 32;
  float sr = br[h], sg = bg[h];
  #pragma unroll
  for (int k = 0; k < 32; ++k) {
    sr += xr[k] * Wr[h * 32 + k];
    sg += xr[k] * Wg[h * 32 + k];
  }
  float ref  = tanhf(sr);
  float gate = 1.f / (1.f + expf(-sg));
  float refined = vout * (1.f - gate) + ref * gate;

  float tp = temps[row & 31];
  tp = tp < TEMP_MINV ? TEMP_MINV : tp;
  Qf2[(size_t)row * 32 + h] = (_Float16)(refined * (LOG2E / tp));
}

// ------------------------- kernel E: final merge ---------------------------
__global__ __launch_bounds__(256) void k_final(
    const float* __restrict__ part, float* __restrict__ out) {
  int t = blockIdx.x * 256 + threadIdx.x;          // NROWS*32 threads
  int row = t >> 5, h = t & 31;
  out[(size_t)row * 32 + h] = merge_out(part, row >> 4, row & 15, h);
}

// ---------------------------------------------------------------------------
extern "C" void kernel_launch(void* const* d_in, const int* in_sizes, int n_in,
                              void* d_out, int out_size, void* d_ws, size_t ws_size,
                              hipStream_t stream) {
  const float* x      = (const float*)d_in[0];
  const float* ve     = (const float*)d_in[1];
  const float* W_ffn  = (const float*)d_in[2];
  const float* b_ffn  = (const float*)d_in[3];
  const float* temps  = (const float*)d_in[4];
  const float* W_ref  = (const float*)d_in[5];
  const float* b_ref  = (const float*)d_in[6];
  const float* W_gate = (const float*)d_in[7];
  const float* b_gate = (const float*)d_in[8];
  float* out = (float*)d_out;

  char* ws = (char*)d_ws;
  _Float16* Mh  = (_Float16*)(ws);                         // 2,048,000 B
  _Float16* Mt  = (_Float16*)(ws + 2048000);               // 2,048,000 B
  _Float16* Qf  = (_Float16*)(ws + 4096000);               //   262,144 B
  _Float16* Qf2 = (_Float16*)(ws + 4358144);               //   262,144 B
  float*    part= (float*)   (ws + 4620288);               // 4,456,448 B

  k_reduce_vocab<<<dim3(VOCAB * 32 / 256), dim3(256), 0, stream>>>(ve, Mh, Mt);
  k_qprep<<<dim3(NROWS * 32 / 256), dim3(256), 0, stream>>>(x, W_ffn, b_ffn, temps, Qf);
  k_flash<<<dim3(GRID_FLASH), dim3(256), 0, stream>>>(Mh, Mt, Qf, part);
  k_merge_refine<<<dim3(NROWS * 32 / 256), dim3(256), 0, stream>>>(
      part, x, W_ref, b_ref, W_gate, b_gate, temps, Qf2);
  k_flash<<<dim3(GRID_FLASH), dim3(256), 0, stream>>>(Mh, Mt, Qf2, part);
  k_final<<<dim3(NROWS * 32 / 256), dim3(256), 0, stream>>>(part, out);
}
